// AttentionLayer_23416161697897
// MI455X (gfx1250) — matrix operand order repaired and checked by reading, not compiled, not run on hardware
//
#include <hip/hip_runtime.h>
#include <hip/hip_bf16.h>
#include <math.h>

// ---------------------------------------------------------------------------
// Attention layer for MI455X (gfx1250, wave32, WMMA + async LDS copies).
//
//   scores = Q K^T / 32, mask -> -inf, A = softmax(scores)      [returned]
//   out1   = A V
//   output = A^T out1                                           [returned]
//
// bf16 conversion pass, then three NT bf16 GEMMs on v_wmma_f32_16x16x32_bf16.
// Block tiles are staged in triple-buffered LDS via
// global_load_async_to_lds_b128 (ASYNCcnt) -> one barrier per k-step, copies
// overlap WMMA math, block-level operand reuse comes from LDS.
// ---------------------------------------------------------------------------

typedef __attribute__((ext_vector_type(16))) __bf16 v16bf;
typedef __attribute__((ext_vector_type(8)))  __bf16 v8bf;
typedef __attribute__((ext_vector_type(4)))  __bf16 v4bf;
typedef __attribute__((ext_vector_type(8)))  float  v8f;

static constexpr int NB  = 16;    // batch
static constexpr int LQS = 2048;  // Lq
static constexpr int SS  = 2048;  // S
static constexpr int DD  = 1024;  // d_model

// --------------------------- fp32 -> bf16 (rowmajor) -----------------------
__global__ __launch_bounds__(256) void k_cvt_bf16(const float* __restrict__ in,
                                                  __bf16* __restrict__ out,
                                                  size_t n4) {
  size_t i = ((size_t)blockIdx.x * blockDim.x + threadIdx.x);
  if (i < n4) {
    const float4 v = reinterpret_cast<const float4*>(in)[i];
    v4bf o;
    o[0] = (__bf16)v.x; o[1] = (__bf16)v.y; o[2] = (__bf16)v.z; o[3] = (__bf16)v.w;
    reinterpret_cast<v4bf*>(out)[i] = o;
  }
}

// --------------------- V[b,k,d] (f32) -> Vt[b,d,k] (bf16) ------------------
__global__ __launch_bounds__(256) void k_cvt_transpose(const float* __restrict__ V,
                                                       __bf16* __restrict__ Vt) {
  __shared__ float tile[32][33];
  const int b  = blockIdx.z;
  const int k0 = blockIdx.y * 32;              // S index
  const int d0 = blockIdx.x * 32;              // D index
  const int tx = threadIdx.x & 31;
  const int ty = threadIdx.x >> 5;             // 0..7
#pragma unroll
  for (int i = ty; i < 32; i += 8)
    tile[i][tx] = V[((size_t)b * SS + (k0 + i)) * DD + d0 + tx];
  __syncthreads();
#pragma unroll
  for (int i = ty; i < 32; i += 8)
    Vt[((size_t)b * DD + (d0 + i)) * SS + k0 + tx] = (__bf16)tile[tx][i];
}

// ---------------------- CDNA5 async-copy / wait helpers --------------------
__device__ __forceinline__ void async_load_b128(unsigned lds_off,
                                                const __bf16* gptr) {
  unsigned long long ga = (unsigned long long)gptr;
  // per-lane: LDS[lds_off .. +15] = MEM[ga .. +15]; tracked with ASYNCcnt
  asm volatile("global_load_async_to_lds_b128 %0, %1, off"
               :: "v"(lds_off), "v"(ga) : "memory");
}
__device__ __forceinline__ void wait_async_le4() {
  asm volatile("s_wait_asynccnt 0x4" ::: "memory");
}
__device__ __forceinline__ void wait_async_0() {
  asm volatile("s_wait_asynccnt 0x0" ::: "memory");
}

// ------------------------------ NT bf16 GEMM -------------------------------
// D[m,n] = sum_k Amat[m,k] * Bmat[n,k]   (both rows K-contiguous)
// MODE 0: scores epilogue  (scale 1/32, mask->-inf, fp32 -> outF[b,m,n])
// MODE 1: out1 epilogue    (bf16, transposed   -> outT[b,n,m])
// MODE 2: plain fp32       (fp32               -> outF[b,m,n])
__device__ __forceinline__ v16bf ld_frag(const __bf16* p0, const __bf16* p1) {
  v8bf lo = *reinterpret_cast<const v8bf*>(p0);
  v8bf hi = *reinterpret_cast<const v8bf*>(p1);
  return __builtin_shufflevector(lo, hi, 0, 1, 2, 3, 4, 5, 6, 7,
                                 8, 9, 10, 11, 12, 13, 14, 15);
}

template <int MODE>
__global__ __launch_bounds__(256) void k_gemm_nt(
    const __bf16* __restrict__ Amat, const __bf16* __restrict__ Bmat,
    int M, int N, int K, int lda, int ldb,
    long long strideA, long long strideB,
    const int* __restrict__ mask, int maskStride,
    float* __restrict__ outF, __bf16* __restrict__ outT) {
  // Triple-buffered 128x32 bf16 block tiles: 3 * (8KB + 8KB) = 48KB LDS.
  __shared__ __bf16 smA[3][128][32];
  __shared__ __bf16 smB[3][128][32];
  constexpr unsigned BUFB = 128 * 32 * 2;  // 8192 bytes per buffer

  const int b = blockIdx.z;
  const __bf16* Ab = Amat + (size_t)b * strideA;
  const __bf16* Bb = Bmat + (size_t)b * strideB;

  const int lane   = threadIdx.x & 31;
  const int l      = lane & 15;   // row-in-tile / N index
  const int h      = lane >> 4;   // K half selector
  const int waveId = threadIdx.x >> 5;
  const int m_blk  = blockIdx.y * 128;
  const int n_blk  = blockIdx.x * 128;
  const int wm     = (waveId & 3) * 32;   // wave tile: 32 (M) x 64 (N)
  const int wn     = (waveId >> 2) * 64;

  // Async staging: 32 wave-ops/k-step (16 A + 16 B), 4 per wave.
  // Each op: 8 rows x 64B; lane -> row (lane>>2), 16B chunk (lane&3).
  // All addressing is loop-invariant except +k0 (global) and +p*BUFB (LDS).
  const int opRow = lane >> 2;
  const int opCol = (lane & 3) * 8;        // element offset within 32-k slab
  const __bf16* srcB[4];
  unsigned dstB[4];
#pragma unroll
  for (int o = 0; o < 4; ++o) {
    const int gi   = waveId * 4 + o;       // 0..31
    const bool isA = gi < 16;
    const int row  = (isA ? gi : gi - 16) * 8 + opRow;
    srcB[o] = isA ? Ab + (size_t)(m_blk + row) * lda + opCol
                  : Bb + (size_t)(n_blk + row) * ldb + opCol;
    dstB[o] = (unsigned)(size_t)(isA ? &smA[0][row][opCol]
                                     : &smB[0][row][opCol]);
  }
  auto issue = [&](int p, int k0) {
    const unsigned poff = (unsigned)p * BUFB;
#pragma unroll
    for (int o = 0; o < 4; ++o)
      async_load_b128(dstB[o] + poff, srcB[o] + k0);
  };

  v8f acc[2][4];
#pragma unroll
  for (int i = 0; i < 2; ++i)
#pragma unroll
    for (int j = 0; j < 4; ++j) acc[i][j] = (v8f){0, 0, 0, 0, 0, 0, 0, 0};

  const int nsteps = K / 32;
  issue(0, 0);
  if (nsteps > 1) issue(1, 32);

  int pc = 0;  // buffer computed this step
  int pi = 2;  // buffer filled for step s+2
  for (int s = 0; s < nsteps; ++s) {
    // group for step s must be complete; group for s+1 may stay in flight
    if (s + 1 < nsteps) wait_async_le4(); else wait_async_0();
    __syncthreads();   // buf pc data visible; buf pi readers (step s-1) done
    if (s + 2 < nsteps) issue(pi, (s + 2) * 32);

    v16bf af[2], bfr[4];
#pragma unroll
    for (int i = 0; i < 2; ++i) {
      // A 16x32 frag: lane (l,h) holds K = h*8..+7 and h*8+16..+23
      const __bf16* pa = &smA[pc][wm + 16 * i + l][h * 8];
      af[i] = ld_frag(pa, pa + 16);
    }
#pragma unroll
    for (int j = 0; j < 4; ++j) {
      // B 32x16 frag: lane (l,h) holds K = h*8..+7 and h*8+16..+23 for column n
      const __bf16* pb = &smB[pc][wn + 16 * j + l][h * 8];
      bfr[j] = ld_frag(pb, pb + 16);
    }
#pragma unroll
    for (int i = 0; i < 2; ++i)
#pragma unroll
      for (int j = 0; j < 4; ++j)
        acc[i][j] = __builtin_amdgcn_wmma_f32_16x16x32_bf16(
            false, af[i], false, bfr[j], (short)0, acc[i][j], false, false);

    pc = (pc == 2) ? 0 : pc + 1;
    pi = (pi == 2) ? 0 : pi + 1;
  }

  // C/D layout: VGPR r -> row (h*8 + r), col l of the 16x16 tile.
#pragma unroll
  for (int i = 0; i < 2; ++i) {
#pragma unroll
    for (int j = 0; j < 4; ++j) {
      const int n  = n_blk + wn + 16 * j + l;
      const int mb = m_blk + wm + 16 * i + h * 8;
      if (MODE == 0) {
        const bool mz = (mask[(size_t)b * maskStride + n] == 0);
#pragma unroll
        for (int r = 0; r < 8; ++r) {
          float v = acc[i][j][r] * 0.03125f;  // 1/sqrt(1024)
          if (mz) v = -__builtin_inff();
          outF[((size_t)b * M + (mb + r)) * N + n] = v;
        }
      } else if (MODE == 1) {
#pragma unroll
        for (int r = 0; r < 8; ++r)
          outT[((size_t)b * N + n) * M + (mb + r)] = (__bf16)acc[i][j][r];
      } else {
#pragma unroll
        for (int r = 0; r < 8; ++r)
          outF[((size_t)b * M + (mb + r)) * N + n] = acc[i][j][r];
      }
    }
  }
}

// ------------------------------ row softmax --------------------------------
__global__ __launch_bounds__(256) void k_softmax(float* __restrict__ scores,
                                                 __bf16* __restrict__ Ab,
                                                 __bf16* __restrict__ ATb) {
  __shared__ float red[256];
  const int row = blockIdx.x;         // b*LQ + q
  const int b   = row / LQS;
  const int q   = row - b * LQS;
  float* sr = scores + (size_t)row * SS;
  const int t = threadIdx.x;

  float vals[8];
  float mx = -__builtin_inff();
#pragma unroll
  for (int i = 0; i < 8; ++i) {
    vals[i] = sr[t + i * 256];
    mx = fmaxf(mx, vals[i]);
  }
  red[t] = mx;
  __syncthreads();
  for (int s = 128; s > 0; s >>= 1) {
    if (t < s) red[t] = fmaxf(red[t], red[t + s]);
    __syncthreads();
  }
  const float rowmax = red[0];
  __syncthreads();

  float sum = 0.f;
#pragma unroll
  for (int i = 0; i < 8; ++i) {
    vals[i] = __expf(vals[i] - rowmax);
    sum += vals[i];
  }
  red[t] = sum;
  __syncthreads();
  for (int s = 128; s > 0; s >>= 1) {
    if (t < s) red[t] += red[t + s];
    __syncthreads();
  }
  const float inv = 1.f / red[0];

#pragma unroll
  for (int i = 0; i < 8; ++i) {
    const int k  = t + i * 256;
    const float a = vals[i] * inv;
    sr[k] = a;
    const __bf16 ab = (__bf16)a;
    Ab[(size_t)row * SS + k]             = ab;
    ATb[((size_t)b * SS + k) * LQS + q]  = ab;
  }
}

// ------------------------------- launcher ----------------------------------
extern "C" void kernel_launch(void* const* d_in, const int* in_sizes, int n_in,
                              void* d_out, int out_size, void* d_ws, size_t ws_size,
                              hipStream_t stream) {
  const float* Q    = (const float*)d_in[0];  // [B, LQ, D]
  const float* Kin  = (const float*)d_in[1];  // [B, S,  D]
  const float* V    = (const float*)d_in[2];  // [B, S,  D]
  const int*   mask = (const int*)d_in[3];    // [B, 1,  S]

  float* outO = (float*)d_out;                       // output  [B, S, D]
  float* outW = outO + (size_t)NB * SS * DD;         // weights [B, LQ, S]

  // Workspace layout (bf16), total 512 MiB.
  char* ws = (char*)d_ws;
  __bf16* Qb  = (__bf16*)ws;  ws += (size_t)NB * LQS * DD * 2;   // 64 MiB
  __bf16* Kb  = (__bf16*)ws;  ws += (size_t)NB * SS  * DD * 2;   // 64 MiB
  __bf16* Vt  = (__bf16*)ws;  ws += (size_t)NB * DD  * SS * 2;   // 64 MiB  [b,d,k]
  __bf16* Ab  = (__bf16*)ws;  ws += (size_t)NB * LQS * SS * 2;   // 128 MiB [b,q,k]
  __bf16* ATb = (__bf16*)ws;  ws += (size_t)NB * SS  * LQS * 2;  // 128 MiB [b,k,q]
  __bf16* Ot  = (__bf16*)ws;  ws += (size_t)NB * DD  * LQS * 2;  // 64 MiB  [b,d,q]
  (void)ws_size; (void)in_sizes; (void)n_in; (void)out_size;

  // 1) precision conversion
  {
    const size_t n4 = (size_t)NB * LQS * DD / 4;
    k_cvt_bf16<<<dim3((unsigned)(n4 / 256)), 256, 0, stream>>>(Q, Qb, n4);
    k_cvt_bf16<<<dim3((unsigned)(n4 / 256)), 256, 0, stream>>>(Kin, Kb, n4);
    k_cvt_transpose<<<dim3(DD / 32, SS / 32, NB), 256, 0, stream>>>(V, Vt);
  }

  // 2) scores = Q K^T / 32 with mask; fp32 into weights region of d_out
  k_gemm_nt<0><<<dim3(SS / 128, LQS / 128, NB), 256, 0, stream>>>(
      Qb, Kb, LQS, SS, DD, DD, DD,
      (long long)LQS * DD, (long long)SS * DD,
      mask, SS, outW, nullptr);

  // 3) softmax in place; emit bf16 A and A^T
  k_softmax<<<dim3(NB * LQS), 256, 0, stream>>>(outW, Ab, ATb);

  // 4) out1 = A V  -> stored transposed bf16 Ot[b,d,q]
  k_gemm_nt<1><<<dim3(DD / 128, LQS / 128, NB), 256, 0, stream>>>(
      Ab, Vt, LQS, DD, SS, SS, SS,
      (long long)LQS * SS, (long long)DD * SS,
      nullptr, 0, nullptr, Ot);

  // 5) output = A^T out1 -> fp32 d_out region 0
  k_gemm_nt<2><<<dim3(DD / 128, SS / 128, NB), 256, 0, stream>>>(
      ATb, Ot, SS, DD, LQS, LQS, LQS,
      (long long)SS * LQS, (long long)DD * LQS,
      nullptr, 0, outO, nullptr);
}
